// Head_30975304138903
// MI455X (gfx1250) — matrix operand order repaired and checked by reading, not compiled, not run on hardware
//
#include <hip/hip_runtime.h>

#define BATCH 16
#define SEQ   2048
#define DIM   64

typedef _Float16 v16h __attribute__((ext_vector_type(16)));
typedef _Float16 v8h  __attribute__((ext_vector_type(8)));
typedef float    v8f  __attribute__((ext_vector_type(8)));

union H16 { v16h v; v8h h[2]; };

// 16 contiguous halves (32B, split into two b128 loads)
__device__ __forceinline__ v16h ld_contig16(const _Float16* p) {
  H16 t; t.h[0] = *(const v8h*)p; t.h[1] = *(const v8h*)(p + 8); return t.v;
}
// two 8-half chunks (A-fragment pattern: K = 8g..8g+7 and 16+8g..16+8g+7)
__device__ __forceinline__ v16h ld_2x8(const _Float16* p0, const _Float16* p1) {
  H16 t; t.h[0] = *(const v8h*)p0; t.h[1] = *(const v8h*)p1; return t.v;
}

__device__ __forceinline__ v8f wmma_f16(v16h a, v16h b, v8f c) {
  // D = A(16x32 f16) x B(32x16 f16) + C(16x16 f32)
  return __builtin_amdgcn_wmma_f32_16x16x32_f16(
      /*neg_a=*/false, a, /*neg_b=*/false, b,
      /*c_mod=*/(short)0, c, /*reuse_a=*/false, /*reuse_b=*/false);
}

// ---------------------------------------------------------------------------
// Phase 1: Q/K/V projections in full fp32, rounded to f16.
//   Qh[b,t,d] = 0.125 * (X @ Wq)   (scale folded in, exact: power of two)
//   Kh[b,t,d] =          X @ Wk
//   Vt[b,d,t] =          X @ Wv    (transposed for contiguous O^T A-fragments)
// ---------------------------------------------------------------------------
__global__ __launch_bounds__(256) void proj_qkv(
    const float* __restrict__ X,  const float* __restrict__ Wq,
    const float* __restrict__ Wk, const float* __restrict__ Wv,
    _Float16* __restrict__ Qh, _Float16* __restrict__ Kh,
    _Float16* __restrict__ Vt) {
  __shared__ float xs[4][DIM];
  const int d  = threadIdx.x;              // 0..63
  const int tr = threadIdx.y;              // 0..3
  const int t  = blockIdx.x * 4 + tr;
  const int b  = blockIdx.y;
  const size_t row = (size_t)(b * SEQ + t) * DIM;
  xs[tr][d] = X[row + d];
  __syncthreads();
  float q = 0.f, k = 0.f, v = 0.f;
#pragma unroll
  for (int i = 0; i < DIM; ++i) {
    const float x = xs[tr][i];
    q = fmaf(x, Wq[i * DIM + d], q);
    k = fmaf(x, Wk[i * DIM + d], k);
    v = fmaf(x, Wv[i * DIM + d], v);
  }
  Qh[row + d] = (_Float16)(0.125f * q);
  Kh[row + d] = (_Float16)k;
  Vt[((size_t)b * DIM + d) * SEQ + t] = (_Float16)v;
}

// ---------------------------------------------------------------------------
// Phase 2: flash attention. 1 wave = 16 queries; 32-key blocks.
//   S^T = K_blk @ Q^T  (softmax stats lane-local), O^T += V^T @ P^T.
// ---------------------------------------------------------------------------
__global__ __launch_bounds__(128) void attn_fwd(
    const _Float16* __restrict__ Qh, const _Float16* __restrict__ Kh,
    const _Float16* __restrict__ Vt, float* __restrict__ Out) {
  __shared__ __align__(16) _Float16 Plds[4][16][32];  // per-wave P tile [m][k]

  const int lane = threadIdx.x & 31;
  const int wave = threadIdx.x >> 5;
  const int m    = lane & 15;   // query column (C/D layout) / fragment row
  const int g    = lane >> 4;   // half-wave group
  const int b    = blockIdx.y;
  const int q0   = (blockIdx.x * 4 + wave) * 16;

  const _Float16* Qb = Qh + (size_t)b * SEQ * DIM;
  const _Float16* Kb = Kh + (size_t)b * SEQ * DIM;
  const _Float16* Vb = Vt + (size_t)b * DIM * SEQ;

  // Q^T B-fragments, resident for the whole row tile.
  // B lane (g,m) holds d = c*32 + {8g..8g+7, 16+8g..16+8g+7} of query row q0+m.
  const _Float16* qp = Qb + (q0 + m) * DIM + g * 8;
  const v16h qt0 = ld_2x8(qp, qp + 16);        // d 0..31 chunk
  const v16h qt1 = ld_2x8(qp + 32, qp + 48);   // d 32..63 chunk

  v8f ot[4];                               // O^T accumulators, 4 d-tiles
#pragma unroll
  for (int dt = 0; dt < 4; ++dt)
#pragma unroll
    for (int r = 0; r < 8; ++r) ot[dt][r] = 0.f;
  float mi = -1e30f, li = 0.f;
  _Float16* pbase = &Plds[wave][0][0];

  for (int j0 = 0; j0 < q0 + 16; j0 += 32) {
    // ---- S^T tiles: A = K rows (lane row = key m of tile), B = Q^T ----
    const _Float16* k0p = Kb + (size_t)(j0 + m) * DIM;
    const _Float16* k1p = Kb + (size_t)(j0 + 16 + m) * DIM;
    v8f s0 = {};
    s0 = wmma_f16(ld_2x8(k0p + g * 8,      k0p + 16 + g * 8), qt0, s0);
    s0 = wmma_f16(ld_2x8(k0p + 32 + g * 8, k0p + 48 + g * 8), qt1, s0);
    v8f s1 = {};
    s1 = wmma_f16(ld_2x8(k1p + g * 8,      k1p + 16 + g * 8), qt0, s1);
    s1 = wmma_f16(ld_2x8(k1p + 32 + g * 8, k1p + 48 + g * 8), qt1, s1);

    // ---- causal mask (only last block is partial) ----
    if (j0 + 31 > q0) {
      const int q = q0 + m;
#pragma unroll
      for (int r = 0; r < 8; ++r) {
        const int kk = j0 + 8 * g + r;       // key of s0[r] in S^T layout
        if (kk > q)      s0[r] = -1e30f;
        if (kk + 16 > q) s1[r] = -1e30f;
      }
    }

    // ---- online softmax: in-lane over 16 vals + one xor-16 combine ----
    float mx = -1e30f;
#pragma unroll
    for (int r = 0; r < 8; ++r) mx = fmaxf(mx, fmaxf(s0[r], s1[r]));
    mx = fmaxf(mx, __shfl_xor(mx, 16, 32));
    const float mn    = fmaxf(mi, mx);
    const float alpha = __expf(mi - mn);
    v8h ph0, ph1;
    float sum = 0.f;
#pragma unroll
    for (int r = 0; r < 8; ++r) { const float p = __expf(s0[r] - mn); sum += p; ph0[r] = (_Float16)p; }
#pragma unroll
    for (int r = 0; r < 8; ++r) { const float p = __expf(s1[r] - mn); sum += p; ph1[r] = (_Float16)p; }
    sum += __shfl_xor(sum, 16, 32);
    li = li * alpha + sum;
    mi = mn;
#pragma unroll
    for (int dt = 0; dt < 4; ++dt)
#pragma unroll
      for (int r = 0; r < 8; ++r) ot[dt][r] *= alpha;

    // ---- transpose P via LDS into B-fragment (P^T) layout ----
    *(v8h*)(pbase + m * 32 +      g * 8) = ph0;   // k = 8g..8g+7
    *(v8h*)(pbase + m * 32 + 16 + g * 8) = ph1;   // k = 16+8g..16+8g+7
    asm volatile("s_wait_dscnt 0" ::: "memory");  // intra-wave LDS RAW fence
    const v16h pt = ld_2x8(pbase + m * 32 + g * 8, pbase + m * 32 + 16 + g * 8);  // k = 8g..8g+7, 16+8g..16+8g+7

    // ---- O^T += V^T @ P^T : A lane row = d-local m of Vt, contiguous cols ----
#pragma unroll
    for (int dt = 0; dt < 4; ++dt) {
      const _Float16* vp = Vb + (size_t)(dt * 16 + m) * SEQ + j0;
      ot[dt] = wmma_f16(ld_2x8(vp + g * 8, vp + 16 + g * 8), pt, ot[dt]);
    }
  }

  // ---- epilogue: O[q0+m][d] = ot/li ; lane holds d = dt*16 + 8g + r ----
  const float inv = 1.0f / li;
  float* orow = Out + ((size_t)(b * SEQ) + q0 + m) * DIM;
#pragma unroll
  for (int dt = 0; dt < 4; ++dt) {
    const float4 f0 = make_float4(ot[dt][0] * inv, ot[dt][1] * inv,
                                  ot[dt][2] * inv, ot[dt][3] * inv);
    const float4 f1 = make_float4(ot[dt][4] * inv, ot[dt][5] * inv,
                                  ot[dt][6] * inv, ot[dt][7] * inv);
    *(float4*)(orow + dt * 16 + g * 8)     = f0;
    *(float4*)(orow + dt * 16 + g * 8 + 4) = f1;
  }
}

extern "C" void kernel_launch(void* const* d_in, const int* in_sizes, int n_in,
                              void* d_out, int out_size, void* d_ws, size_t ws_size,
                              hipStream_t stream) {
  (void)in_sizes; (void)n_in; (void)out_size; (void)ws_size;
  const float* X  = (const float*)d_in[0];
  // d_in[1] = bool causal mask — unused, causality is computed analytically.
  const float* Wq = (const float*)d_in[2];
  const float* Wk = (const float*)d_in[3];
  const float* Wv = (const float*)d_in[4];
  float* Out = (float*)d_out;

  _Float16* Qh = (_Float16*)d_ws;                         // 4 MB
  _Float16* Kh = Qh + (size_t)BATCH * SEQ * DIM;          // 4 MB
  _Float16* Vt = Kh + (size_t)BATCH * SEQ * DIM;          // 4 MB (transposed)

  proj_qkv<<<dim3(SEQ / 4, BATCH), dim3(64, 4), 0, stream>>>(X, Wq, Wk, Wv, Qh, Kh, Vt);
  attn_fwd<<<dim3(SEQ / 64, BATCH), dim3(128), 0, stream>>>(Qh, Kh, Vt, Out);
}
